// CustomHGTModel_34368328303045
// MI455X (gfx1250) — matrix operand order repaired and checked by reading, not compiled, not run on hardware
//
#include <hip/hip_runtime.h>
#include <hip/hip_bf16.h>

typedef __attribute__((ext_vector_type(16))) __bf16 v16bf;
typedef __attribute__((ext_vector_type(8)))  float  v8f;
typedef __attribute__((ext_vector_type(4)))  float  v4f;
typedef __attribute__((ext_vector_type(2)))  float  v2f;

#define N_NODES 100000

// ---------------------------------------------------------------------------
// Count edges per destination node (mean denominator; same for both layers).
// ---------------------------------------------------------------------------
__global__ void hgt_count_kernel(const int* __restrict__ dst,
                                 float* __restrict__ cnt, int E) {
  int e = blockIdx.x * blockDim.x + threadIdx.x;
  if (e < E) atomicAdd(&cnt[dst[e]], 1.0f);
}

// ---------------------------------------------------------------------------
// Fused gather -> (16x66)@(66x64) WMMA -> ReLU -> atomic scatter-add.
// One wave per 16-edge tile.
//   K 0..63  : gathered node rows, bf16, 2x v_wmma_f32_16x16x32_bf16 / N-tile
//   K 64..65 : [edge_attr, 1] x [W64; bias], fp32, 1x v_wmma_f32_16x16x4_f32
// B fragments hoisted out of the edge loop; C starts from inline 0; edge
// metadata for the next tile software-pipelined behind the gathers.
// ---------------------------------------------------------------------------
__global__ __launch_bounds__(256) void hgt_edge_layer_kernel(
    const float* __restrict__ x,     // [N, 64] input node features
    const int*   __restrict__ ei,    // [2, E]  edge index (row0=src, row1=dst)
    const float* __restrict__ attr,  // [E]
    const float* __restrict__ W,     // [65, 64]
    const float* __restrict__ b,     // [64]
    float*       __restrict__ acc,   // [N, 64] zero-initialized accumulator
    int E) {
  __shared__ __align__(32) __bf16 sWt[64 * 64];  // W[0:64,:] transposed: [n][k]
  __shared__ float sW64[64];                     // W[64,:] (edge_attr row)
  __shared__ float sB[64];                       // bias

  for (int idx = threadIdx.x; idx < 65 * 64; idx += 256) {
    int k = idx >> 6, n = idx & 63;
    float v = W[idx];
    if (k < 64) sWt[n * 64 + k] = (__bf16)v;
    else        sW64[n] = v;
  }
  if (threadIdx.x < 64) sB[threadIdx.x] = b[threadIdx.x];
  __syncthreads();

  const int lane = threadIdx.x & 31;
  const int wave = threadIdx.x >> 5;
  const int n0   = lane & 15;   // column within a 16-wide N tile / edge row id
  const int half = lane >> 4;   // lane half selects K sub-ranges

  // Persistent bf16 B fragments: 4 N-tiles x 2 K-chunks (ds_load_b128 pairs).
  v16bf Bf[4][2];
#pragma unroll
  for (int j = 0; j < 4; ++j) {
#pragma unroll
    for (int kc = 0; kc < 2; ++kc) {
      const int n = n0 + 16 * j;
      const int kbase = kc * 32 + 16 * half;
      Bf[j][kc] = *(const v16bf*)(sWt + n * 64 + kbase);
    }
  }
  // Persistent f32 B fragments for the K=4 tail: rows {W64, bias, 0, 0}.
  // f32 4x16 B layout: lane half 0 holds {B[0][n], B[1][n]}, half 1 {B[2][n], B[3][n]}.
  v2f Bf32[4];
#pragma unroll
  for (int j = 0; j < 4; ++j) {
    const int n = n0 + 16 * j;
    Bf32[j][0] = half ? 0.0f : sW64[n];
    Bf32[j][1] = half ? 0.0f : sB[n];
  }
  const float oneh = half ? 0.0f : 1.0f;  // constant-1 feature, half 0 only

  const int ntiles = E >> 4;                // E = 1M -> 62500 tiles
  const int gwave  = blockIdx.x * 8 + wave; // 8 waves per 256-thread block
  const int nwaves = gridDim.x * 8;

  if (gwave >= ntiles) return;              // wave-uniform exit, EXEC stays full

  // Prologue: edge metadata for the first tile.
  int   e    = gwave * 16 + n0;             // lanes m and m+16 share edge m
  int   src  = ei[e];
  int   dstn = ei[E + e];
  float ae   = attr[e];

  for (int tile = gwave; tile < ntiles; tile += nwaves) {
    // -------- issue all 8 gather loads up front ----------------------------
    const float* xr = x + (size_t)src * 64;
    v4f f[2][4];
#pragma unroll
    for (int kc = 0; kc < 2; ++kc) {
      const int o = kc * 32 + 16 * half;    // A 16x32 bf16 layout K sub-ranges
      f[kc][0] = *(const v4f*)(xr + o);
      f[kc][1] = *(const v4f*)(xr + o + 4);
      f[kc][2] = *(const v4f*)(xr + o + 8);
      f[kc][3] = *(const v4f*)(xr + o + 12);
    }

    // -------- prefetch next tile's edge metadata behind the gathers --------
    const int next = tile + nwaves;
    const int pe   = (next < ntiles ? next : tile) * 16 + n0;  // clamp, no branch
    const int   nsrc = ei[pe];
    const int   ndst = ei[E + pe];
    const float nae  = attr[pe];

    // -------- per-output-row dst offsets (C VGPR r holds row m = r+8*half) -
    int rowoff[8];
#pragma unroll
    for (int r = 0; r < 8; ++r)
      rowoff[r] = __shfl(dstn, r + 8 * half, 32) << 6;  // 32-bit elem offset

    // -------- f32 A tail fragment: columns [attr, 1] -----------------------
    // f32 16x4 A layout: lane half 0 holds {A[m][0], A[m][1]}, half 1 {A[m][2], A[m][3]}.
    v2f a32;
    a32[0] = half ? 0.0f : ae;
    a32[1] = oneh;

    // -------- convert gathered rows to bf16 A fragments --------------------
    v16bf Af[2];
#pragma unroll
    for (int kc = 0; kc < 2; ++kc) {
      v16bf a;
#pragma unroll
      for (int t = 0; t < 4; ++t) {
        a[t]      = (__bf16)f[kc][0][t];
        a[4 + t]  = (__bf16)f[kc][1][t];
        a[8 + t]  = (__bf16)f[kc][2][t];
        a[12 + t] = (__bf16)f[kc][3][t];
      }
      Af[kc] = a;
    }

    // -------- 4 N-tiles: 1 f32 K=4 WMMA + 2 bf16 K=32 WMMAs, ReLU, scatter -
#pragma unroll
    for (int j = 0; j < 4; ++j) {
      const int n = n0 + 16 * j;
      v8f c = {};  // inline-0 C
      c = __builtin_amdgcn_wmma_f32_16x16x4_f32(false, a32, false, Bf32[j],
                                                (short)0, c, false, false);
      c = __builtin_amdgcn_wmma_f32_16x16x32_bf16(false, Af[0], false, Bf[j][0],
                                                  (short)0, c, false, false);
      c = __builtin_amdgcn_wmma_f32_16x16x32_bf16(false, Af[1], false, Bf[j][1],
                                                  (short)0, c, false, false);
#pragma unroll
      for (int r = 0; r < 8; ++r) {
        const float m = fmaxf(c[r], 0.0f);           // relu(msg)
        atomicAdd(acc + rowoff[r] + n, m);           // saddr + 32-bit voffset
      }
    }

    src = nsrc; dstn = ndst; ae = nae;
  }
}

// ---------------------------------------------------------------------------
// acc[i] = relu(acc[i] / max(cnt[node], 1))  (outer ReLU of each layer)
// ---------------------------------------------------------------------------
__global__ void hgt_finalize_kernel(float* __restrict__ acc,
                                    const float* __restrict__ cnt, int total) {
  int i = blockIdx.x * blockDim.x + threadIdx.x;
  if (i < total) {
    float d = fmaxf(cnt[i >> 6], 1.0f);
    acc[i] = fmaxf(acc[i] / d, 0.0f);
  }
}

extern "C" void kernel_launch(void* const* d_in, const int* in_sizes, int n_in,
                              void* d_out, int out_size, void* d_ws, size_t ws_size,
                              hipStream_t stream) {
  const float* x    = (const float*)d_in[0];
  const int*   ei   = (const int*)d_in[1];
  const float* attr = (const float*)d_in[2];
  const float* w1   = (const float*)d_in[3];
  const float* b1   = (const float*)d_in[4];
  const float* w2   = (const float*)d_in[5];
  const float* b2   = (const float*)d_in[6];
  float* out = (float*)d_out;

  const int E = in_sizes[2];  // edge_attr is [E]
  const int N = N_NODES;
  const size_t feat = (size_t)N * 64;

  float* cnt = (float*)d_ws;   // N floats
  float* h   = cnt + N;        // N*64 floats (layer-1 activations)

  hipMemsetAsync(cnt, 0, (size_t)N * sizeof(float), stream);
  hipMemsetAsync(h,   0, feat * sizeof(float), stream);
  hipMemsetAsync(out, 0, feat * sizeof(float), stream);

  hgt_count_kernel<<<(E + 255) / 256, 256, 0, stream>>>(ei + E, cnt, E);

  // Layer 1: x -> h
  hgt_edge_layer_kernel<<<1024, 256, 0, stream>>>(x, ei, attr, w1, b1, h, E);
  hgt_finalize_kernel<<<((int)feat + 255) / 256, 256, 0, stream>>>(h, cnt, (int)feat);

  // Layer 2: h -> out
  hgt_edge_layer_kernel<<<1024, 256, 0, stream>>>(h, ei, attr, w2, b2, out, E);
  hgt_finalize_kernel<<<((int)feat + 255) / 256, 256, 0, stream>>>(out, cnt, (int)feat);
}
